// AnomalyTransformer_6863357739045
// MI455X (gfx1250) — matrix operand order repaired and checked by reading, not compiled, not run on hardware
//
#include <hip/hip_runtime.h>
#include <hip/hip_bf16.h>
#include <math.h>

typedef __attribute__((ext_vector_type(16))) __bf16 v16bf;
typedef __attribute__((ext_vector_type(8)))  __bf16 v8bf;
typedef __attribute__((ext_vector_type(8)))  float  v8f;

#define B_    8
#define L_    512
#define ENC_  38
#define COUT_ 38
#define D_    512
#define H_    8
#define DH_   64
#define DFF_  512
#define NL_   3

// ---------------------------------------------------------------------------
// Generic bf16 WMMA GEMM:  D = act(alpha*(A * Bt^T) + bias + R), written to
// fp32 C and/or bf16 Cb with arbitrary row/col strides (lets us emit the V
// matrix pre-transposed, fuse residual adds, GELU, and bf16 conversion).
// A is MxK row-major (row stride lda), Bt is NxK row-major (row stride ldb),
// i.e. Bt = B^T. Batch (grid.z): z1 = z/zdiv, z2 = z%zdiv with separate
// strides, covering both plain batching and (b,head) batching.
// Block = 256 threads = 8 waves; each wave does a 16x32 tile via two
// v_wmma_f32_16x16x32_bf16 per K-step; block tile = 64x64.
// ---------------------------------------------------------------------------
struct GemmP {
  const __bf16* A;  long long sA1, sA2;  int lda;
  const __bf16* Bt; long long sB1, sB2;  int ldb;
  const float* bias;                                  // per-n, may be null
  const float* R;   long long sR1, sR2;  int ldrm, ldrn;   // residual, may be null
  float*  C;        long long sC1, sC2;  int ldcm, ldcn;   // fp32 out, may be null
  __bf16* Cb;       long long sCb1, sCb2; int ldcbm, ldcbn; // bf16 out, may be null
  int M, N, K, zdiv;
  float alpha;
  int act;                                            // 0 = none, 1 = exact GELU
};

__device__ __forceinline__ v16bf cat8(v8bf lo, v8bf hi) {
  return __builtin_shufflevector(lo, hi, 0,1,2,3,4,5,6,7,8,9,10,11,12,13,14,15);
}

__global__ __launch_bounds__(256) void gemm_bf16_wmma(GemmP p) {
  const int zd = (p.zdiv > 0) ? p.zdiv : 1;
  const int z  = blockIdx.z;
  const int z1 = z / zd, z2 = z % zd;

  const __bf16* A  = p.A  + z1 * p.sA1 + z2 * p.sA2;
  const __bf16* Bt = p.Bt + z1 * p.sB1 + z2 * p.sB2;

  const int w    = threadIdx.x >> 5;   // wave 0..7
  const int lane = threadIdx.x & 31;
  const int m0 = blockIdx.y * 64 + (w & 3) * 16;
  const int n0 = blockIdx.x * 64 + (w >> 2) * 32;
  if (m0 >= p.M || n0 >= p.N) return;

  v8f acc0 = {}, acc1 = {};

  // A fragment: row = m0 + lane%16; contiguous K half per lane-half (ko=0/16)
  const int arow = m0 + (lane & 15);
  const int ko   = (lane < 16) ? 0 : 16;
  const __bf16* Ab = A + (long long)arow * p.lda + ko;
  // B fragment: n = n0 + lane%16; contiguous K half per lane-half (kb=0/16)
  const int brow = n0 + (lane & 15);
  const int kb   = (lane < 16) ? 0 : 16;
  const __bf16* B0 = Bt + (long long)brow * p.ldb + kb;
  const __bf16* B1 = B0 + (long long)16 * p.ldb;

  for (int k0 = 0; k0 < p.K; k0 += 32) {
    v16bf a  = cat8(*(const v8bf*)(Ab + k0), *(const v8bf*)(Ab + k0 + 8));
    v16bf b0 = cat8(*(const v8bf*)(B0 + k0), *(const v8bf*)(B0 + k0 + 8));
    v16bf b1 = cat8(*(const v8bf*)(B1 + k0), *(const v8bf*)(B1 + k0 + 8));
    acc0 = __builtin_amdgcn_wmma_f32_16x16x32_bf16(false, a, false, b0,
                                                   (short)0, acc0, false, false);
    acc1 = __builtin_amdgcn_wmma_f32_16x16x32_bf16(false, a, false, b1,
                                                   (short)0, acc1, false, false);
  }

  // Epilogue. C layout: lanes 0-15 -> N=lane, M=vgpr; lanes 16-31 -> N=lane-16, M=vgpr+8
  const int mbase = m0 + ((lane >> 4) << 3);
  const long long offR  = p.R  ? (z1 * p.sR1  + z2 * p.sR2)  : 0;
  const long long offC  = p.C  ? (z1 * p.sC1  + z2 * p.sC2)  : 0;
  const long long offCb = p.Cb ? (z1 * p.sCb1 + z2 * p.sCb2) : 0;

  for (int t = 0; t < 2; ++t) {
    v8f acc = t ? acc1 : acc0;
    const int n = n0 + (lane & 15) + t * 16;
    const float bia = p.bias ? p.bias[n] : 0.0f;
    for (int i = 0; i < 8; ++i) {
      const int m = mbase + i;
      float v = p.alpha * acc[i] + bia;
      if (p.R) v += p.R[offR + (long long)m * p.ldrm + (long long)n * p.ldrn];
      if (p.act == 1) v = 0.5f * v * (1.0f + erff(v * 0.70710678118654752f));
      if (p.C)  p.C [offC  + (long long)m * p.ldcm  + (long long)n * p.ldcn ] = v;
      if (p.Cb) p.Cb[offCb + (long long)m * p.ldcbm + (long long)n * p.ldcbn] = (__bf16)v;
    }
  }
}

// ---------------------------------------------------------------------------
// LDS-tiled transpose + fp32->bf16 convert: WT[n*K+k] = bf16(W[k*N+n])
// ---------------------------------------------------------------------------
__global__ __launch_bounds__(256) void transpose_bf16_kernel(
    const float* __restrict__ W, __bf16* __restrict__ WT, int K, int N) {
  __shared__ float tile[32][33];
  const int kb = blockIdx.x * 32, nb = blockIdx.y * 32;
  const int tx = threadIdx.x & 31, ty = threadIdx.x >> 5;
  for (int i = ty; i < 32; i += 8)
    tile[i][tx] = W[(long long)(kb + i) * N + nb + tx];
  __syncthreads();
  for (int i = ty; i < 32; i += 8)
    WT[(long long)(nb + i) * K + kb + tx] = (__bf16)tile[tx][i];
}

// ---------------------------------------------------------------------------
// Token embedding (kernel-3 circular conv over ENC_IN) + sinusoidal PE.
// One thread per (b,l,d); writes fp32 h and bf16 hb.
// ---------------------------------------------------------------------------
__global__ __launch_bounds__(256) void embed_kernel(
    const float* __restrict__ x, const float* __restrict__ tokw,
    float* __restrict__ h, __bf16* __restrict__ hb) {
  const long long idx = (long long)blockIdx.x * 256 + threadIdx.x; // B*L*D
  const int d  = (int)(idx % D_);
  const long long bl = idx / D_;
  const int l  = (int)(bl % L_);
  const int b  = (int)(bl / L_);
  float acc = 0.0f;
  for (int kh = 0; kh < 3; ++kh) {
    int r = l + kh - 1;
    r = (r < 0) ? (L_ - 1) : ((r >= L_) ? 0 : r);
    const float* xr = x + ((long long)b * L_ + r) * ENC_;
    const float* wr = tokw + (long long)kh * ENC_ * D_ + d;
    for (int i = 0; i < ENC_; ++i) acc += xr[i] * wr[(long long)i * D_];
  }
  const int m2 = d & ~1;
  const float div = expf((float)m2 * (-9.210340371976184f / (float)D_)); // -ln(1e4)/D
  const float ang = (float)l * div;
  acc += (d & 1) ? cosf(ang) : sinf(ang);
  h[idx]  = acc;
  hb[idx] = (__bf16)acc;
}

// ---------------------------------------------------------------------------
// sigma head: s = 3^(sigmoid(5*(h.Wsig+b)) + 1e-5) - 1, stored (b,l,head)
// ---------------------------------------------------------------------------
__global__ __launch_bounds__(256) void sig_kernel(
    const float* __restrict__ h, const float* __restrict__ Wsig,
    const float* __restrict__ bsig, float* __restrict__ sval) {
  const int idx = blockIdx.x * 256 + threadIdx.x;      // B*L*H
  if (idx >= B_ * L_ * H_) return;
  const int hd = idx % H_;
  const long long bl = idx / H_;
  const float* hr = h + bl * D_;
  float acc = bsig[hd];
  for (int d = 0; d < D_; ++d) acc += hr[d] * Wsig[(long long)d * H_ + hd];
  float s = 1.0f / (1.0f + expf(-5.0f * acc)) + 1e-5f;
  s = powf(3.0f, s) - 1.0f;
  sval[idx] = s;
}

// ---------------------------------------------------------------------------
// Gaussian prior + broadcast sigma. One block per (b,head,l) row of 512 cols.
// ---------------------------------------------------------------------------
__global__ __launch_bounds__(128) void prior_kernel(
    const float* __restrict__ sval, float* __restrict__ prior,
    float* __restrict__ sigma) {
  const long long r = blockIdx.x;                      // B*H*L rows
  const int l  = (int)(r % L_);
  const long long bh = r / L_;
  const int hd = (int)(bh % H_);
  const int b  = (int)(bh / H_);
  const float s = sval[((long long)b * L_ + l) * H_ + hd];
  const float c0 = -1.0f / (2.0f * s * s);
  const float c1 = 0.3989422804014327f / s;            // 1/(sqrt(2*pi)*s)
  const long long base = r * L_;
  for (int j = threadIdx.x; j < L_; j += 128) {
    const float d = (float)(l - j);
    prior[base + j] = expf(d * d * c0) * c1;
    sigma[base + j] = s;
  }
}

// ---------------------------------------------------------------------------
// In-place row softmax over 512 cols (wave per row, 16 elems/lane),
// also emitting a bf16 copy for the series.V WMMA GEMM.
// ---------------------------------------------------------------------------
__global__ __launch_bounds__(256) void softmax_kernel(
    float* __restrict__ S, __bf16* __restrict__ Sb, float scale) {
  const int w = threadIdx.x >> 5, lane = threadIdx.x & 31;
  const long long row = (long long)blockIdx.x * 8 + w;
  float* p  = S  + row * L_;
  __bf16* pb = Sb + row * L_;
  float vals[16];
  float mx = -3.4e38f;
  for (int i = 0; i < 16; ++i) {
    vals[i] = p[i * 32 + lane] * scale;
    mx = fmaxf(mx, vals[i]);
  }
  for (int o = 16; o > 0; o >>= 1) mx = fmaxf(mx, __shfl_xor(mx, o, 32));
  float sum = 0.0f;
  for (int i = 0; i < 16; ++i) { vals[i] = expf(vals[i] - mx); sum += vals[i]; }
  for (int o = 16; o > 0; o >>= 1) sum += __shfl_xor(sum, o, 32);
  const float inv = 1.0f / sum;
  for (int i = 0; i < 16; ++i) {
    const float v = vals[i] * inv;
    p [i * 32 + lane] = v;
    pb[i * 32 + lane] = (__bf16)v;
  }
}

// ---------------------------------------------------------------------------
// LayerNorm over last dim (512). Wave per row; optional bf16 copy out.
// ---------------------------------------------------------------------------
__global__ __launch_bounds__(256) void layernorm_kernel(
    const float* __restrict__ in, const float* __restrict__ g,
    const float* __restrict__ bt, float* __restrict__ out,
    __bf16* __restrict__ outb) {
  const int w = threadIdx.x >> 5, lane = threadIdx.x & 31;
  const long long row = (long long)blockIdx.x * 8 + w;
  const float* p = in + row * D_;
  float vals[16];
  float s = 0.0f;
  for (int i = 0; i < 16; ++i) { vals[i] = p[i * 32 + lane]; s += vals[i]; }
  for (int o = 16; o > 0; o >>= 1) s += __shfl_xor(s, o, 32);
  const float mean = s * (1.0f / D_);
  float vs = 0.0f;
  for (int i = 0; i < 16; ++i) { const float d = vals[i] - mean; vs += d * d; }
  for (int o = 16; o > 0; o >>= 1) vs += __shfl_xor(vs, o, 32);
  const float inv = rsqrtf(vs * (1.0f / D_) + 1e-5f);
  for (int i = 0; i < 16; ++i) {
    const int c = i * 32 + lane;
    const float v = g[c] * (vals[i] - mean) * inv + bt[c];
    out[row * D_ + c] = v;
    if (outb) outb[row * D_ + c] = (__bf16)v;
  }
}

// ---------------------------------------------------------------------------
// Final projection: out[b,l,c] = hf[b,l,:] . Wp[:,c] + bp[c]  (C_OUT=38)
// ---------------------------------------------------------------------------
__global__ __launch_bounds__(256) void proj_kernel(
    const float* __restrict__ hf, const float* __restrict__ Wp,
    const float* __restrict__ bp, float* __restrict__ out) {
  const int idx = blockIdx.x * 256 + threadIdx.x;      // B*L*COUT
  if (idx >= B_ * L_ * COUT_) return;
  const int c = idx % COUT_;
  const long long bl = idx / COUT_;
  const float* hr = hf + bl * D_;
  float acc = bp[c];
  for (int d = 0; d < D_; ++d) acc += hr[d] * Wp[(long long)d * COUT_ + c];
  out[idx] = acc;
}

// ===========================================================================
extern "C" void kernel_launch(void* const* d_in, const int* in_sizes, int n_in,
                              void* d_out, int out_size, void* d_ws, size_t ws_size,
                              hipStream_t stream) {
  (void)in_sizes; (void)n_in; (void)out_size; (void)ws_size;

  const float* x    = (const float*)d_in[0];
  const float* tokw = (const float*)d_in[1];
  const float* norm_g = (const float*)d_in[56];
  const float* norm_b = (const float*)d_in[57];
  const float* Wp     = (const float*)d_in[58];
  const float* bp     = (const float*)d_in[59];
  float* outp = (float*)d_out;

  const long long BLD  = (long long)B_ * L_ * D_;        // 2,097,152
  const long long SEG  = (long long)B_ * H_ * L_ * L_;   // 16,777,216
  const long long OUTN = (long long)B_ * L_ * COUT_;     // 155,648

  // -------- workspace carve --------
  char* cur = (char*)d_ws;
  auto carve = [&](size_t bytes) -> void* {
    void* p = (void*)cur;
    cur += (bytes + 255) & ~(size_t)255;
    return p;
  };
  float*  h    = (float*) carve(BLD * 4);
  float*  h2   = (float*) carve(BLD * 4);
  float*  x1   = (float*) carve(BLD * 4);
  __bf16* hb   = (__bf16*)carve(BLD * 2);
  __bf16* qb   = (__bf16*)carve(BLD * 2);
  __bf16* kb   = (__bf16*)carve(BLD * 2);
  __bf16* vTb  = (__bf16*)carve(BLD * 2);   // (B, D, L)
  __bf16* Vb   = (__bf16*)carve(BLD * 2);   // (B, L, D)
  __bf16* x1b  = (__bf16*)carve(BLD * 2);
  __bf16* yb   = (__bf16*)carve(BLD * 2);
  __bf16* sbf  = (__bf16*)carve(SEG * 2);   // bf16 series for series.V GEMM
  float*  sval = (float*) carve((size_t)B_ * L_ * H_ * 4);
  __bf16* WqT  = (__bf16*)carve((size_t)D_ * D_ * 2);
  __bf16* WkT  = (__bf16*)carve((size_t)D_ * D_ * 2);
  __bf16* WvT  = (__bf16*)carve((size_t)D_ * D_ * 2);
  __bf16* WoT  = (__bf16*)carve((size_t)D_ * D_ * 2);
  __bf16* W1T  = (__bf16*)carve((size_t)D_ * DFF_ * 2);
  __bf16* W2T  = (__bf16*)carve((size_t)DFF_ * D_ * 2);

  // -------- embedding + PE --------
  embed_kernel<<<(int)(BLD / 256), 256, 0, stream>>>(x, tokw, h, hb);

  const dim3 tb(256);
  const dim3 gT(D_ / 32, D_ / 32);
  const float scale = 0.125f;  // 1/sqrt(D/H)

  for (int li = 0; li < NL_; ++li) {
    const int base = 2 + li * 18;
    const float* Wq = (const float*)d_in[base + 0];
    const float* bq = (const float*)d_in[base + 1];
    const float* Wk = (const float*)d_in[base + 2];
    const float* bk = (const float*)d_in[base + 3];
    const float* Wv = (const float*)d_in[base + 4];
    const float* bv = (const float*)d_in[base + 5];
    const float* Wsig = (const float*)d_in[base + 6];
    const float* bsig = (const float*)d_in[base + 7];
    const float* Wo = (const float*)d_in[base + 8];
    const float* bo = (const float*)d_in[base + 9];
    const float* W1 = (const float*)d_in[base + 10];
    const float* bf1 = (const float*)d_in[base + 11];
    const float* W2 = (const float*)d_in[base + 12];
    const float* bf2 = (const float*)d_in[base + 13];
    const float* g1 = (const float*)d_in[base + 14];
    const float* b1 = (const float*)d_in[base + 15];
    const float* g2 = (const float*)d_in[base + 16];
    const float* b2 = (const float*)d_in[base + 17];

    float* seriesO = outp + OUTN + (long long)li * SEG;
    float* priorO  = outp + OUTN + (long long)(3 + li) * SEG;
    float* sigmaO  = outp + OUTN + (long long)(6 + li) * SEG;

    // weight transpose+convert to bf16 (Bt layout for the WMMA GEMM)
    transpose_bf16_kernel<<<gT, tb, 0, stream>>>(Wq, WqT, D_, D_);
    transpose_bf16_kernel<<<gT, tb, 0, stream>>>(Wk, WkT, D_, D_);
    transpose_bf16_kernel<<<gT, tb, 0, stream>>>(Wv, WvT, D_, D_);
    transpose_bf16_kernel<<<gT, tb, 0, stream>>>(Wo, WoT, D_, D_);
    transpose_bf16_kernel<<<gT, tb, 0, stream>>>(W1, W1T, D_, DFF_);
    transpose_bf16_kernel<<<gT, tb, 0, stream>>>(W2, W2T, DFF_, D_);

    // sigma head (reads pre-layer h) -> s; prior + sigma outputs
    sig_kernel<<<(B_ * L_ * H_ + 255) / 256, tb, 0, stream>>>(h, Wsig, bsig, sval);
    prior_kernel<<<B_ * H_ * L_, 128, 0, stream>>>(sval, priorO, sigmaO);

    // Q = hb.Wq + bq -> bf16 (B*L, D)
    GemmP pq = {};
    pq.A = hb; pq.lda = D_; pq.Bt = WqT; pq.ldb = D_; pq.bias = bq;
    pq.Cb = qb; pq.ldcbm = D_; pq.ldcbn = 1;
    pq.M = B_ * L_; pq.N = D_; pq.K = D_; pq.zdiv = 1; pq.alpha = 1.0f;
    gemm_bf16_wmma<<<dim3(D_ / 64, (B_ * L_) / 64, 1), tb, 0, stream>>>(pq);

    // K = hb.Wk + bk
    GemmP pk = pq; pk.Bt = WkT; pk.bias = bk; pk.Cb = kb;
    gemm_bf16_wmma<<<dim3(D_ / 64, (B_ * L_) / 64, 1), tb, 0, stream>>>(pk);

    // V = hb.Wv + bv, emitted TRANSPOSED per batch: vT[b][d][l]
    GemmP pv = {};
    pv.A = hb; pv.sA1 = (long long)L_ * D_; pv.lda = D_;
    pv.Bt = WvT; pv.ldb = D_; pv.bias = bv;
    pv.Cb = vTb; pv.sCb1 = (long long)D_ * L_; pv.ldcbm = 1; pv.ldcbn = L_;
    pv.M = L_; pv.N = D_; pv.K = D_; pv.zdiv = 1; pv.alpha = 1.0f;
    gemm_bf16_wmma<<<dim3(D_ / 64, L_ / 64, B_), tb, 0, stream>>>(pv);

    // scores[b,h] = Q[b,:,h,:] . K[b,:,h,:]^T  -> fp32 straight into d_out
    GemmP ps = {};
    ps.A = qb;  ps.sA1 = (long long)L_ * D_; ps.sA2 = DH_; ps.lda = D_;
    ps.Bt = kb; ps.sB1 = (long long)L_ * D_; ps.sB2 = DH_; ps.ldb = D_;
    ps.C = seriesO; ps.sC1 = (long long)H_ * L_ * L_; ps.sC2 = (long long)L_ * L_;
    ps.ldcm = L_; ps.ldcn = 1;
    ps.M = L_; ps.N = L_; ps.K = DH_; ps.zdiv = H_; ps.alpha = 1.0f;
    gemm_bf16_wmma<<<dim3(L_ / 64, L_ / 64, B_ * H_), tb, 0, stream>>>(ps);

    // series = softmax(scale*scores) in place (+ bf16 copy)
    softmax_kernel<<<(B_ * H_ * L_) / 8, tb, 0, stream>>>(seriesO, sbf, scale);

    // V' = series . V  -> Vb (B, L, D) bf16
    GemmP pV = {};
    pV.A = sbf; pV.sA1 = (long long)H_ * L_ * L_; pV.sA2 = (long long)L_ * L_; pV.lda = L_;
    pV.Bt = vTb; pV.sB1 = (long long)D_ * L_; pV.sB2 = (long long)DH_ * L_; pV.ldb = L_;
    pV.Cb = Vb; pV.sCb1 = (long long)L_ * D_; pV.sCb2 = DH_; pV.ldcbm = D_; pV.ldcbn = 1;
    pV.M = L_; pV.N = DH_; pV.K = L_; pV.zdiv = H_; pV.alpha = 1.0f;
    gemm_bf16_wmma<<<dim3(1, L_ / 64, B_ * H_), tb, 0, stream>>>(pV);

    // h2 = h + Vb.Wo + bo
    GemmP po = {};
    po.A = Vb; po.lda = D_; po.Bt = WoT; po.ldb = D_; po.bias = bo;
    po.R = h; po.ldrm = D_; po.ldrn = 1;
    po.C = h2; po.ldcm = D_; po.ldcn = 1;
    po.M = B_ * L_; po.N = D_; po.K = D_; po.zdiv = 1; po.alpha = 1.0f;
    gemm_bf16_wmma<<<dim3(D_ / 64, (B_ * L_) / 64, 1), tb, 0, stream>>>(po);

    // x1 = LN(h2)
    layernorm_kernel<<<(B_ * L_) / 8, tb, 0, stream>>>(h2, g1, b1, x1, x1b);

    // y = gelu(x1.W1 + bf1) -> bf16
    GemmP p1 = {};
    p1.A = x1b; p1.lda = D_; p1.Bt = W1T; p1.ldb = D_; p1.bias = bf1;
    p1.Cb = yb; p1.ldcbm = DFF_; p1.ldcbn = 1;
    p1.M = B_ * L_; p1.N = DFF_; p1.K = D_; p1.zdiv = 1; p1.alpha = 1.0f; p1.act = 1;
    gemm_bf16_wmma<<<dim3(DFF_ / 64, (B_ * L_) / 64, 1), tb, 0, stream>>>(p1);

    // t = x1 + y.W2 + bf2   (into h2)
    GemmP p2 = {};
    p2.A = yb; p2.lda = DFF_; p2.Bt = W2T; p2.ldb = DFF_; p2.bias = bf2;
    p2.R = x1; p2.ldrm = D_; p2.ldrn = 1;
    p2.C = h2; p2.ldcm = D_; p2.ldcn = 1;
    p2.M = B_ * L_; p2.N = D_; p2.K = DFF_; p2.zdiv = 1; p2.alpha = 1.0f;
    gemm_bf16_wmma<<<dim3(D_ / 64, (B_ * L_) / 64, 1), tb, 0, stream>>>(p2);

    // h = LN(t)   (fp32 + bf16 for next layer's QKV)
    layernorm_kernel<<<(B_ * L_) / 8, tb, 0, stream>>>(h2, g2, b2, h, hb);
  }

  // final norm + projection
  layernorm_kernel<<<(B_ * L_) / 8, tb, 0, stream>>>(h, norm_g, norm_b, x1, nullptr);
  proj_kernel<<<(int)((OUTN + 255) / 256), tb, 0, stream>>>(x1, Wp, bp, outp);
}
